// GENE_NET_27298812133390
// MI455X (gfx1250) — compile-verified, matrix operand order checked by reading, not run on hardware
//
#include <hip/hip_runtime.h>
#include <hip/hip_bf16.h>
#include <stdint.h>

typedef __attribute__((ext_vector_type(16))) _Float16 v16h;
typedef __attribute__((ext_vector_type(4)))  _Float16 h4;
typedef __attribute__((ext_vector_type(8)))  float    v8f;

#define BN_INV 0.9999950000374997f

__device__ __forceinline__ float gelu(float x) {
    return 0.5f * x * (1.0f + erff(x * 0.7071067811865475f));
}

// K-index packing inside a 16-bit 16x32 WMMA fragment (ISA 7.12.2):
// lanes 0-15 hold K 0..7 (v0..v3) / 16..23 (v4..v7); lanes 16-31 hold 8..15 / 24..31.
__device__ __forceinline__ int frag_k(int e, int half) {
    int j = e >> 1, r = e & 1;
    return (j < 4) ? (half * 8 + 2 * j + r) : (16 + half * 8 + 2 * (j - 4) + r);
}

__device__ __forceinline__ v8f wmma16(v16h a, v16h b, v8f c) {
    return __builtin_amdgcn_wmma_f32_16x16x32_f16(false, a, false, b, (short)0, c, false, false);
}

// CDNA5 async copy global->LDS (ASYNCcnt path). LDS offset = low 32 bits of flat shared ptr.
__device__ __forceinline__ void async_b128(uint32_t ldsOff, const float* g) {
    asm volatile("global_load_async_to_lds_b128 %0, %1, off" :: "v"(ldsOff), "v"(g) : "memory");
}
__device__ __forceinline__ void wait_async_le4() {
    asm volatile("s_wait_asynccnt 0x4" ::: "memory");
}
__device__ __forceinline__ void wait_async_le0() {
    asm volatile("s_wait_asynccnt 0x0" ::: "memory");
}

// ---------------------------------------------------------------- leaf stage
__global__ __launch_bounds__(128)
void k_leaf(const float* __restrict__ x, const int* __restrict__ gene_idx,
            const float* __restrict__ W1, const float* __restrict__ b1,
            const float* __restrict__ W2, const float* __restrict__ b2,
            const float* __restrict__ g, const float* __restrict__ bt,
            const float* __restrict__ aux0W, const float* __restrict__ aux0b,
            float* __restrict__ h0, float* __restrict__ aux0_out) {
    __shared__ float h[8][16];
    __shared__ float u[8][64];
    const int t = blockIdx.x, tid = threadIdx.x;
    { int b = tid >> 4, d = tid & 15; h[b][d] = x[b * 2048 + gene_idx[t * 16 + d]]; }
    __syncthreads();
    for (int i = 0; i < 2; ++i) {
        for (int p = tid; p < 512; p += 128) {
            int b = p >> 6, hc = p & 63;
            float s = b1[(i * 128 + t) * 64 + hc];
            for (int d = 0; d < 16; ++d)
                s += h[b][d] * W1[((size_t)((i * 128 + t) * 16 + d)) * 64 + hc];
            u[b][hc] = gelu(s);
        }
        __syncthreads();
        int b = tid >> 4, d = tid & 15;
        float s = b2[(i * 128 + t) * 16 + d];
        for (int hc = 0; hc < 64; ++hc)
            s += u[b][hc] * W2[((size_t)((i * 128 + t) * 64 + hc)) * 16 + d];
        float hn = s * BN_INV * g[(i * 128 + t) * 16 + d] + bt[(i * 128 + t) * 16 + d] + h[b][d];
        __syncthreads();
        h[b][d] = hn;
        __syncthreads();
    }
    { int b = tid >> 4, d = tid & 15; h0[((size_t)b * 128 + t) * 16 + d] = h[b][d]; }
    if (tid < 8) {
        float s = aux0b[t];
        for (int d = 0; d < 16; ++d) s += h[tid][d] * aux0W[t * 16 + d];
        aux0_out[tid * 128 + t] = s;
    }
}

__global__ void k_gather(const float* __restrict__ h0, const int* __restrict__ child_idx,
                         float* __restrict__ h1in) {
    int idx = blockIdx.x * blockDim.x + threadIdx.x;
    if (idx >= 8 * 16 * 128) return;
    int b = idx >> 11, rem = idx & 2047;
    int t = rem >> 7, cd = rem & 127;
    int c = cd >> 4, d = cd & 15;
    h1in[idx] = h0[((size_t)b * 128 + child_idx[t * 8 + c]) * 16 + d];
}

// ---------------------------------------------------------------- mid stage (WMMA per term)
__global__ __launch_bounds__(256)
void k_mid(const float* __restrict__ h1in, const float* __restrict__ W1,
           const float* __restrict__ b1, const float* __restrict__ W2,
           const float* __restrict__ b2, const float* __restrict__ g,
           const float* __restrict__ bt, const float* __restrict__ aux1W,
           const float* __restrict__ aux1b, float* __restrict__ r,
           float* __restrict__ aux1_out) {
    __shared__ float hbuf[8][128];
    __shared__ __align__(16) _Float16 Aab[16][128];
    __shared__ __align__(16) _Float16 ubuf[16][512];
    const int t = blockIdx.x, tid = threadIdx.x;
    const int lane = tid & 31, wave = tid >> 5, half = lane >> 4, lm = lane & 15;
    for (int p = tid; p < 1024; p += 256) {
        int b = p >> 7, c = p & 127;
        hbuf[b][c] = h1in[((size_t)b * 16 + t) * 128 + c];
    }
    __syncthreads();
    for (int i = 0; i < 2; ++i) {
        for (int p = tid; p < 2048; p += 256) {
            int m = p >> 7, c = p & 127;
            Aab[m][c] = (_Float16)(m < 8 ? hbuf[m][c] : 0.f);
        }
        __syncthreads();
        const float* W1t = W1 + ((size_t)(i * 16 + t)) * 128 * 512;
        for (int qd = 0; qd < 4; ++qd) {                 // 8 waves * 4 tiles = 512 cols
            int col0 = (wave * 4 + qd) * 16;
            v8f acc = {};
            for (int k0 = 0; k0 < 128; k0 += 32) {
                v16h a, bf;
                for (int e = 0; e < 16; ++e) {
                    int kk = frag_k(e, half);
                    a[e]  = Aab[lm][k0 + kk];
                    bf[e] = (_Float16)W1t[(size_t)(k0 + kk) * 512 + col0 + lm];
                }
                acc = wmma16(a, bf, acc);
            }
            for (int vv = 0; vv < 8; ++vv) {
                int m = vv + 8 * half, col = col0 + lm;
                ubuf[m][col] = (_Float16)gelu(acc[vv] + b1[(i * 16 + t) * 512 + col]);
            }
        }
        __syncthreads();
        const float* W2t = W2 + ((size_t)(i * 16 + t)) * 512 * 128;
        {
            int col0 = wave * 16;
            v8f acc = {};
            for (int k0 = 0; k0 < 512; k0 += 32) {
                v16h a, bf;
                for (int e = 0; e < 16; ++e) {
                    int kk = frag_k(e, half);
                    a[e]  = ubuf[lm][k0 + kk];
                    bf[e] = (_Float16)W2t[(size_t)(k0 + kk) * 128 + col0 + lm];
                }
                acc = wmma16(a, bf, acc);
            }
            __syncthreads();
            for (int vv = 0; vv < 8; ++vv) {
                int m = vv + 8 * half, col = col0 + lm;
                if (m < 8) {
                    float val = (acc[vv] + b2[(i * 16 + t) * 128 + col]) * BN_INV *
                                    g[(i * 16 + t) * 128 + col] +
                                bt[(i * 16 + t) * 128 + col] + hbuf[m][col];
                    hbuf[m][col] = val;
                }
            }
        }
        __syncthreads();
    }
    for (int p = tid; p < 1024; p += 256) {
        int b = p >> 7, c = p & 127;
        r[(size_t)b * 2048 + t * 128 + c] = hbuf[b][c];
    }
    if (tid < 8) {
        float s = aux1b[t];
        for (int c = 0; c < 128; ++c) s += hbuf[tid][c] * aux1W[t * 128 + c];
        aux1_out[tid * 16 + t] = s;
    }
}

// ------------------------------------------------ skinny GEMM (M=8 pad 16), double-buffered
// async global->LDS weight pipeline: issue tile s+1, s_wait_asynccnt 4 drains tile s (in-order).
// MODE 0: out = gelu(A@W + bias)
// MODE 1: out = (A@W + bias)*BN_INV*g + bt + resid
// MODE 2: out = A@W + bias
template <int MODE>
__global__ __launch_bounds__(256)
void k_gemm(const float* __restrict__ A, const float* __restrict__ W,
            const float* __restrict__ bias, const float* __restrict__ g,
            const float* __restrict__ bt, const float* __restrict__ resid,
            float* __restrict__ out, int K, int H) {
    __shared__ __align__(16) _Float16 Abuf[16 * 1024];
    __shared__ __align__(16) float Wbuf[2][32 * 128];
    const int tid = threadIdx.x, lane = tid & 31, wave = tid >> 5;
    const int half = lane >> 4, lm = lane & 15;
    const int col0 = blockIdx.x * 128 + wave * 16;
    const int nsteps = K >> 5;
    v8f acc = {};

    {   // issue W tile for step 0 into buffer 0 (4 b128 per thread)
        const float* wsrc = W + (size_t)blockIdx.x * 128;
        const uint32_t wb = (uint32_t)(uintptr_t)(&Wbuf[0][0]);
        for (int c = tid; c < 1024; c += 256) {
            int rr = c >> 5, cc = (c & 31) * 4;
            async_b128(wb + (uint32_t)(rr * 128 + cc) * 4u, wsrc + (size_t)rr * H + cc);
        }
    }
    for (int s = 0; s < nsteps; ++s) {
        const int k0 = s << 5;
        if ((k0 & 1023) == 0) {
            // restage next 1024-wide A chunk as f16 (M pad to 16), float4 -> packed h4
            for (int p = tid; p < 16 * 256; p += 256) {
                int m = p >> 8, kk4 = (p & 255) * 4;
                h4 hv = {};
                if (m < 8) {
                    float4 v4 = *(const float4*)(A + (size_t)m * K + k0 + kk4);
                    hv[0] = (_Float16)v4.x; hv[1] = (_Float16)v4.y;
                    hv[2] = (_Float16)v4.z; hv[3] = (_Float16)v4.w;
                }
                *(h4*)(&Abuf[m * 1024 + kk4]) = hv;
            }
        }
        if (s + 1 < nsteps) {   // issue-ahead: tile s+1 into the other buffer
            const float* wsrc = W + (size_t)(k0 + 32) * H + blockIdx.x * 128;
            const uint32_t wb = (uint32_t)(uintptr_t)(&Wbuf[(s + 1) & 1][0]);
            for (int c = tid; c < 1024; c += 256) {
                int rr = c >> 5, cc = (c & 31) * 4;
                async_b128(wb + (uint32_t)(rr * 128 + cc) * 4u, wsrc + (size_t)rr * H + cc);
            }
            __builtin_prefetch(wsrc + (size_t)32 * H, 0, 1);   // warm L2 two tiles ahead
            wait_async_le4();
        } else {
            wait_async_le0();
        }
        __syncthreads();
        const float* wb = Wbuf[s & 1];
        const int ka = k0 & 1023;
        v16h a, bf;
        for (int e = 0; e < 16; ++e) {
            int kk = frag_k(e, half);
            a[e]  = Abuf[lm * 1024 + ka + kk];
            bf[e] = (_Float16)wb[kk * 128 + wave * 16 + lm];
        }
        acc = wmma16(a, bf, acc);
        __syncthreads();
    }
    for (int vv = 0; vv < 8; ++vv) {
        int m = vv + 8 * half, col = col0 + lm;
        if (m < 8 && col < H) {
            float val = acc[vv] + bias[col];
            if (MODE == 0) val = gelu(val);
            if (MODE == 1) val = val * BN_INV * g[col] + bt[col] + resid[(size_t)m * H + col];
            out[(size_t)m * H + col] = val;
        }
    }
}

__global__ __launch_bounds__(256)
void k_auxr(const float* __restrict__ r, const float* __restrict__ w,
            const float* __restrict__ b, float* __restrict__ outp) {
    int wave = threadIdx.x >> 5, lane = threadIdx.x & 31;
    float s = 0.f;
    for (int c = lane; c < 2048; c += 32) s += r[(size_t)wave * 2048 + c] * w[c];
    for (int d = 16; d >= 1; d >>= 1) s += __shfl_xor(s, d, 32);
    if (lane == 0) outp[wave] = s + b[0];
}

// ---------------------------------------------------------------- side branch
__global__ void k_copy(const float* __restrict__ a, float* __restrict__ o, int n) {
    int i = blockIdx.x * blockDim.x + threadIdx.x;
    if (i < n) o[i] = a[i];
}
__global__ void k_add(const float* __restrict__ a, const float* __restrict__ b,
                      float* __restrict__ o, int n) {
    int i = blockIdx.x * blockDim.x + threadIdx.x;
    if (i < n) o[i] = a[i] + b[i];
}

// qkv stored as (b, n, h*8+d): contiguous 64 floats per point -> float4 stores,
// and attention rows stay contiguous 8-float segments (float4 loads).
__global__ void k_side_qkv(const float* __restrict__ sideh, const float* __restrict__ Wqkv,
                           int i, float* __restrict__ q, float* __restrict__ k,
                           float* __restrict__ v) {
    int idx = blockIdx.x * blockDim.x + threadIdx.x;
    if (idx >= 16384) return;
    float x = sideh[idx];
    const float* wq = Wqkv + i * 192;
    float4* qo = (float4*)(q + (size_t)idx * 64);
    float4* ko = (float4*)(k + (size_t)idx * 64);
    float4* vo = (float4*)(v + (size_t)idx * 64);
    for (int c4 = 0; c4 < 16; ++c4) {
        int hd = c4 * 4;
        qo[c4] = make_float4(x * wq[hd], x * wq[hd + 1], x * wq[hd + 2], x * wq[hd + 3]);
        ko[c4] = make_float4(x * wq[64 + hd], x * wq[64 + hd + 1],
                             x * wq[64 + hd + 2], x * wq[64 + hd + 3]);
        vo[c4] = make_float4(x * wq[128 + hd], x * wq[128 + hd + 1],
                             x * wq[128 + hd + 2], x * wq[128 + hd + 3]);
    }
}

// Flash attention, d=8 (K padded to 32): only the low half-wave carries K=0..7,
// so Q/K fragments are two b128 loads on lanes 0-15 and nothing on lanes 16-31.
__global__ __launch_bounds__(128)
void k_side_attn(const float* __restrict__ q, const float* __restrict__ k,
                 const float* __restrict__ v, const float* __restrict__ mask,
                 float* __restrict__ o) {
    __shared__ __align__(16) _Float16 pbuf[4][16 * 32];
    const int tid = threadIdx.x, lane = tid & 31, wave = tid >> 5;
    const int half = lane >> 4, lm = lane & 15;
    const int bh = blockIdx.x;                    // b*8+h
    const int b = bh >> 3, hh = bh & 7;
    const int r0 = blockIdx.y * 64 + wave * 16;   // 16-row tile per wave
    const float scale = 0.35355339059327373f;
    const float* Q  = q + ((size_t)b * 2048) * 64 + hh * 8;
    const float* Kp = k + ((size_t)b * 2048) * 64 + hh * 8;
    const float* Vp = v + ((size_t)b * 2048) * 64 + hh * 8;
    v16h aq = {};
    if (half == 0) {
        const float* qp = Q + (size_t)(r0 + lm) * 64;
        float4 q0 = *(const float4*)qp, q1 = *(const float4*)(qp + 4);
        aq[0] = (_Float16)q0.x; aq[1] = (_Float16)q0.y;
        aq[2] = (_Float16)q0.z; aq[3] = (_Float16)q0.w;
        aq[4] = (_Float16)q1.x; aq[5] = (_Float16)q1.y;
        aq[6] = (_Float16)q1.z; aq[7] = (_Float16)q1.w;
    }
    v8f co = {};
    float mrow[8], lrow[8];
    for (int t = 0; t < 8; ++t) { mrow[t] = -1e30f; lrow[t] = 0.f; }
    for (int jc = 0; jc < 2048; jc += 32) {
        v16h b0 = {}, b1 = {};
        if (half == 0) {
            const float* kp0 = Kp + (size_t)(jc + lm) * 64;
            const float* kp1 = Kp + (size_t)(jc + 16 + lm) * 64;
            float4 a0 = *(const float4*)kp0, a1 = *(const float4*)(kp0 + 4);
            float4 c0 = *(const float4*)kp1, c1 = *(const float4*)(kp1 + 4);
            b0[0] = (_Float16)a0.x; b0[1] = (_Float16)a0.y;
            b0[2] = (_Float16)a0.z; b0[3] = (_Float16)a0.w;
            b0[4] = (_Float16)a1.x; b0[5] = (_Float16)a1.y;
            b0[6] = (_Float16)a1.z; b0[7] = (_Float16)a1.w;
            b1[0] = (_Float16)c0.x; b1[1] = (_Float16)c0.y;
            b1[2] = (_Float16)c0.z; b1[3] = (_Float16)c0.w;
            b1[4] = (_Float16)c1.x; b1[5] = (_Float16)c1.y;
            b1[6] = (_Float16)c1.z; b1[7] = (_Float16)c1.w;
        }
        v8f z = {};
        v8f s0 = wmma16(aq, b0, z);
        v8f s1 = wmma16(aq, b1, z);
        for (int t = 0; t < 8; ++t) {
            int row = r0 + t + 8 * half;
            float x0 = s0[t] * scale * mask[(size_t)row * 2048 + jc + lm];
            float x1 = s1[t] * scale * mask[(size_t)row * 2048 + jc + 16 + lm];
            float mx = fmaxf(x0, x1);
            for (int d = 8; d >= 1; d >>= 1) mx = fmaxf(mx, __shfl_xor(mx, d, 32));
            float newm = fmaxf(mrow[t], mx);
            float corr = __expf(mrow[t] - newm);
            float p0 = __expf(x0 - newm), p1 = __expf(x1 - newm);
            float ps = p0 + p1;
            for (int d = 8; d >= 1; d >>= 1) ps += __shfl_xor(ps, d, 32);
            lrow[t] = lrow[t] * corr + ps;
            mrow[t] = newm;
            co[t] *= corr;
            pbuf[wave][(t + 8 * half) * 32 + lm]      = (_Float16)p0;
            pbuf[wave][(t + 8 * half) * 32 + 16 + lm] = (_Float16)p1;
        }
        __syncthreads();
        v16h ap, bv = {};
        for (int e = 0; e < 16; ++e) {
            int kk = frag_k(e, half);
            ap[e] = pbuf[wave][lm * 32 + kk];
            if (lm < 8) bv[e] = (_Float16)Vp[(size_t)(jc + kk) * 64 + lm];
        }
        co = wmma16(ap, bv, co);
        __syncthreads();
    }
    for (int t = 0; t < 8; ++t) {
        int row = r0 + t + 8 * half;
        if (lm < 8)
            o[(((size_t)b * 2048 + row)) * 64 + hh * 8 + lm] = co[t] / lrow[t];
    }
}

__global__ void k_side_post(const float* __restrict__ o, const float* __restrict__ Wo,
                            const float* __restrict__ bo, const float* __restrict__ FW1,
                            const float* __restrict__ Fb1, const float* __restrict__ FW2,
                            const float* __restrict__ Fb2, const float* __restrict__ sg,
                            const float* __restrict__ sbt, int i, float* __restrict__ sideh) {
    int idx = blockIdx.x * blockDim.x + threadIdx.x;
    if (idx >= 16384) return;
    float hv = sideh[idx];
    float s = bo[i];
    const float4* orow = (const float4*)(o + (size_t)idx * 64);
    for (int j4 = 0; j4 < 16; ++j4) {
        float4 ov = orow[j4];
        const float* wo = Wo + i * 64 + j4 * 4;
        s += ov.x * wo[0] + ov.y * wo[1] + ov.z * wo[2] + ov.w * wo[3];
    }
    hv = s + hv;
    float s2 = Fb2[i];
    for (int c = 0; c < 4; ++c) {
        float uu = gelu(hv * FW1[i * 4 + c] + Fb1[i * 4 + c]);
        s2 += uu * FW2[i * 4 + c];
    }
    hv = s2 * BN_INV * sg[i] + sbt[i] + hv;
    sideh[idx] = hv;
}

// ---------------------------------------------------------------- host orchestration
extern "C" void kernel_launch(void* const* d_in, const int* in_sizes, int n_in,
                              void* d_out, int out_size, void* d_ws, size_t ws_size,
                              hipStream_t stream) {
    (void)in_sizes; (void)n_in; (void)out_size; (void)ws_size;
    const float* x        = (const float*)d_in[0];
    const int*   gene_idx = (const int*)d_in[1];
    const int*   child_idx= (const int*)d_in[2];
    const float* mask     = (const float*)d_in[3];
    const float *lW1 = (const float*)d_in[4],  *lb1 = (const float*)d_in[5],
                *lW2 = (const float*)d_in[6],  *lb2 = (const float*)d_in[7],
                *lg  = (const float*)d_in[8],  *lbt = (const float*)d_in[9];
    const float *mW1 = (const float*)d_in[10], *mb1 = (const float*)d_in[11],
                *mW2 = (const float*)d_in[12], *mb2 = (const float*)d_in[13],
                *mg  = (const float*)d_in[14], *mbt = (const float*)d_in[15];
    const float *rW1 = (const float*)d_in[16], *rb1 = (const float*)d_in[17],
                *rW2 = (const float*)d_in[18], *rb2 = (const float*)d_in[19],
                *rg  = (const float*)d_in[20], *rbt = (const float*)d_in[21];
    const float *aux0W = (const float*)d_in[22], *aux0b = (const float*)d_in[23];
    const float *aux1W = (const float*)d_in[24], *aux1b = (const float*)d_in[25];
    const float *auxrW = (const float*)d_in[26], *auxrb = (const float*)d_in[27];
    const float *fW1 = (const float*)d_in[28], *fb1 = (const float*)d_in[29],
                *fW2 = (const float*)d_in[30], *fb2 = (const float*)d_in[31],
                *fg  = (const float*)d_in[32], *fbt = (const float*)d_in[33];
    const float *faW = (const float*)d_in[34], *fab = (const float*)d_in[35];
    const float *Wqkv = (const float*)d_in[36], *Wo = (const float*)d_in[37],
                *bo   = (const float*)d_in[38], *FW1 = (const float*)d_in[39],
                *Fb1  = (const float*)d_in[40], *FW2 = (const float*)d_in[41],
                *Fb2  = (const float*)d_in[42], *sg  = (const float*)d_in[43],
                *sbt  = (const float*)d_in[44], *outW = (const float*)d_in[45],
                *outb = (const float*)d_in[46];

    float* ws      = (float*)d_ws;
    float* h0      = ws;                 // 16384
    float* h1in    = ws + 16384;         // 16384
    float* r       = ws + 32768;         // 16384
    float* u       = ws + 49152;         // 65536
    float* f       = ws + 114688;        // 16384
    float* sideh   = ws + 131072;        // 16384
    float* qb      = ws + 147456;        // 1048576
    float* kb      = qb + 1048576;
    float* vb      = kb + 1048576;
    float* ob      = vb + 1048576;
    float* sideout = ob + 1048576;       // 16384
    float* tmp     = sideout + 16384;    // 16384  (~17.5 MB total)

    float* outF  = (float*)d_out;        // final (8,128)
    float* outA0 = outF + 1024;          // aux0  (8,128)
    float* outA1 = outA0 + 1024;         // aux1  (8,16)
    float* outAr = outA1 + 128;          // auxr  (8,)

    k_leaf<<<128, 128, 0, stream>>>(x, gene_idx, lW1, lb1, lW2, lb2, lg, lbt,
                                    aux0W, aux0b, h0, outA0);
    k_gather<<<64, 256, 0, stream>>>(h0, child_idx, h1in);
    k_mid<<<16, 256, 0, stream>>>(h1in, mW1, mb1, mW2, mb2, mg, mbt,
                                  aux1W, aux1b, r, outA1);
    for (int i = 0; i < 2; ++i) {
        k_gemm<0><<<64, 256, 0, stream>>>(r, rW1 + (size_t)i * 2048 * 8192, rb1 + i * 8192,
                                          nullptr, nullptr, nullptr, u, 2048, 8192);
        k_gemm<1><<<16, 256, 0, stream>>>(u, rW2 + (size_t)i * 8192 * 2048, rb2 + i * 2048,
                                          rg + i * 2048, rbt + i * 2048, r, r, 8192, 2048);
    }
    k_auxr<<<1, 256, 0, stream>>>(r, auxrW, auxrb, outAr);
    k_gemm<0><<<64, 256, 0, stream>>>(r, fW1, fb1, nullptr, nullptr, nullptr, u, 2048, 8192);
    k_gemm<1><<<16, 256, 0, stream>>>(u, fW2, fb2, fg, fbt, r, f, 8192, 2048);

    k_copy<<<64, 256, 0, stream>>>(x, sideh, 16384);
    for (int i = 0; i < 2; ++i) {
        k_side_qkv<<<64, 256, 0, stream>>>(sideh, Wqkv, i, qb, kb, vb);
        k_side_attn<<<dim3(64, 32), 128, 0, stream>>>(qb, kb, vb, mask, ob);
        k_side_post<<<64, 256, 0, stream>>>(ob, Wo, bo, FW1, Fb1, FW2, Fb2, sg, sbt, i, sideh);
    }
    k_gemm<2><<<16, 256, 0, stream>>>(sideh, outW, outb, nullptr, nullptr, nullptr,
                                      sideout, 2048, 2048);
    k_add<<<64, 256, 0, stream>>>(f, sideout, tmp, 16384);
    k_gemm<2><<<1, 256, 0, stream>>>(tmp, faW, fab, nullptr, nullptr, nullptr,
                                     outF, 2048, 128);
}
